// Transformer_Pseudoformer_86844238725648
// MI455X (gfx1250) — matrix operand order repaired and checked by reading, not compiled, not run on hardware
//
#include <hip/hip_runtime.h>
#include <hip/hip_bf16.h>
#include <math.h>

// ---------------- problem dims ----------------
constexpr int NB  = 16;
constexpr int NL  = 1024;
constexpr int NC  = 512;
constexpr int NP  = 64;
constexpr int NH  = 8;
constexpr int NDK = 64;
constexpr int HDK = NH * NDK;   // 512
constexpr int FF  = 4 * NC;     // 2048

typedef __attribute__((ext_vector_type(16))) _Float16 v16h;
typedef __attribute__((ext_vector_type(8)))  float    v8f;

union FragU  { unsigned u[8]; v16h v; };
union PackU  { _Float16 h[4]; uint2 u2; };

__device__ __forceinline__ int imin(int a, int b) { return a < b ? a : b; }

__device__ __forceinline__ float gelu_exact(float x) {
    return 0.5f * x * (1.0f + erff(x * 0.70710678118654752f));
}

// ---------------- WMMA GEMM ----------------
// C[z] = act( alpha * op(A[z]) * op(B[z]) + bias )
// Assumptions (hold for every launch below): K % 32 == 0, N % 64 == 0,
// M ragged allowed (guarded via clamp+select on loads, guard on stores).
struct GemmP {
    const float* A; const float* Bm; const float* bias; float* C;
    int M, N, K;
    int lda, ldb, ldc;
    long long sA0, sA1, sB0, sB1, sC0, sC1;
    int subBatch;
    float alpha;
    int act; // 0=none 1=gelu
};

#define BMT 128
#define BNT 64
#define BKT 32
#define LDP (BKT + 4)   // LDS row pitch in halves: 36 halves = 72 B (8B aligned)

template <int TA, int TB>
__global__ __launch_bounds__(256) void gemm_wmma_f16(GemmP pr) {
    __shared__ __align__(16) _Float16 lA[BMT][LDP];
    __shared__ __align__(16) _Float16 lB[BNT][LDP];

    const int z  = blockIdx.z;
    const int zb = z / pr.subBatch, zh = z % pr.subBatch;
    const float* A  = pr.A  + (long long)zb * pr.sA0 + (long long)zh * pr.sA1;
    const float* Bm = pr.Bm + (long long)zb * pr.sB0 + (long long)zh * pr.sB1;
    float*       C  = pr.C  + (long long)zb * pr.sC0 + (long long)zh * pr.sC1;

    const int m0 = blockIdx.x * BMT;
    const int n0 = blockIdx.y * BNT;
    const int tid  = threadIdx.x;
    const int lane = tid & 31;
    const int wid  = tid >> 5;
    const int wm = (wid >> 1) * 32;   // 4 wave rows
    const int wn = (wid & 1) * 32;    // 2 wave cols

    v8f acc[2][2];
    #pragma unroll
    for (int i = 0; i < 2; ++i)
        #pragma unroll
        for (int j = 0; j < 2; ++j)
            #pragma unroll
            for (int e = 0; e < 8; ++e) acc[i][j][e] = 0.0f;

    const int r15 = lane & 15;
    const int hi  = lane >> 4;

    float4 aReg[4];
    float4 bReg[2];

    // ---- global fetch of one K-tile into registers (vectorized b128) ----
    auto fetchA = [&](int kt) {
        #pragma unroll
        for (int i = 0; i < 4; ++i) {
            int vidx = i * 256 + tid;
            if (TA == 0) {
                // A[m][k] row-major, contiguous k. 8 float4 per row.
                int m = vidx >> 3, kk = (vidx & 7) * 4;
                int gm = m0 + m;
                const float4* src = (const float4*)(A + (long long)imin(gm, pr.M - 1) * pr.lda + kt + kk);
                float4 v = *src;
                if (gm >= pr.M) v = make_float4(0.f, 0.f, 0.f, 0.f);
                aReg[i] = v;
            } else {
                // A physical [k][m], contiguous m. 32 float4 per k-row.
                int k = vidx >> 5, mv = (vidx & 31) * 4;
                int gm = m0 + mv;
                const float4* src = (const float4*)(A + (long long)(kt + k) * pr.lda + imin(gm, pr.M - 4));
                float4 v = *src;
                if (gm >= pr.M) v = make_float4(0.f, 0.f, 0.f, 0.f);
                aReg[i] = v;
            }
        }
    };
    auto fetchB = [&](int kt) {
        #pragma unroll
        for (int i = 0; i < 2; ++i) {
            int vidx = i * 256 + tid;
            if (TB == 0) {
                // B physical [k][n], contiguous n. 16 float4 per k-row.
                int k = vidx >> 4, nv = (vidx & 15) * 4;
                bReg[i] = *(const float4*)(Bm + (long long)(kt + k) * pr.ldb + n0 + nv);
            } else {
                // B physical [n][k], contiguous k. 8 float4 per n-row.
                int n = vidx >> 3, kv = (vidx & 7) * 4;
                bReg[i] = *(const float4*)(Bm + (long long)(n0 + n) * pr.ldb + kt + kv);
            }
        }
    };
    // ---- convert + store registers into LDS tiles ----
    auto stageA = [&]() {
        #pragma unroll
        for (int i = 0; i < 4; ++i) {
            int vidx = i * 256 + tid;
            PackU p;
            p.h[0] = (_Float16)aReg[i].x; p.h[1] = (_Float16)aReg[i].y;
            p.h[2] = (_Float16)aReg[i].z; p.h[3] = (_Float16)aReg[i].w;
            if (TA == 0) {
                int m = vidx >> 3, kk = (vidx & 7) * 4;
                *(uint2*)&lA[m][kk] = p.u2;           // contiguous k: one b64 store
            } else {
                int k = vidx >> 5, mv = (vidx & 31) * 4;
                lA[mv + 0][k] = p.h[0]; lA[mv + 1][k] = p.h[1];
                lA[mv + 2][k] = p.h[2]; lA[mv + 3][k] = p.h[3];
            }
        }
    };
    auto stageB = [&]() {
        #pragma unroll
        for (int i = 0; i < 2; ++i) {
            int vidx = i * 256 + tid;
            PackU p;
            p.h[0] = (_Float16)bReg[i].x; p.h[1] = (_Float16)bReg[i].y;
            p.h[2] = (_Float16)bReg[i].z; p.h[3] = (_Float16)bReg[i].w;
            if (TB == 0) {
                int k = vidx >> 4, nv = (vidx & 15) * 4;
                lB[nv + 0][k] = p.h[0]; lB[nv + 1][k] = p.h[1];
                lB[nv + 2][k] = p.h[2]; lB[nv + 3][k] = p.h[3];
            } else {
                int n = vidx >> 3, kv = (vidx & 7) * 4;
                *(uint2*)&lB[n][kv] = p.u2;           // contiguous k: one b64 store
            }
        }
    };

    // ---- software-pipelined main loop ----
    fetchA(0);
    fetchB(0);
    for (int kt = 0; kt < pr.K; kt += BKT) {
        stageA();
        stageB();
        __syncthreads();

        if (kt + BKT < pr.K) {   // overlap next global fetch with ds+wmma
            fetchA(kt + BKT);
            fetchB(kt + BKT);
        }

        v16h afr[2], bfr[2];
        #pragma unroll
        for (int im = 0; im < 2; ++im) {
            int row = wm + im * 16 + r15;
            FragU u;
            #pragma unroll
            for (int i = 0; i < 8; ++i) {
                int kk = ((i < 4) ? 0 : 16) + hi * 8 + 2 * (i & 3);
                u.u[i] = *(const unsigned*)&lA[row][kk];
            }
            afr[im] = u.v;
        }
        #pragma unroll
        for (int in = 0; in < 2; ++in) {
            int col = wn + in * 16 + r15;
            FragU u;
            #pragma unroll
            for (int i = 0; i < 8; ++i) {
                int kk = ((i < 4) ? 0 : 16) + hi * 8 + 2 * (i & 3);
                u.u[i] = *(const unsigned*)&lB[col][kk];
            }
            bfr[in] = u.v;
        }
        #pragma unroll
        for (int im = 0; im < 2; ++im)
            #pragma unroll
            for (int in = 0; in < 2; ++in)
                acc[im][in] = __builtin_amdgcn_wmma_f32_16x16x32_f16(
                    false, afr[im], false, bfr[in],
                    (short)0, acc[im][in], false, false);
        __syncthreads();
    }

    // ---- store D: VGPR rr -> M = rr + 8*hi ; N = lane&15 ----
    #pragma unroll
    for (int im = 0; im < 2; ++im)
        #pragma unroll
        for (int in = 0; in < 2; ++in)
            #pragma unroll
            for (int rr = 0; rr < 8; ++rr) {
                int m = m0 + wm + im * 16 + hi * 8 + rr;
                int n = n0 + wn + in * 16 + r15;
                if (m < pr.M) {
                    float v = acc[im][in][rr] * pr.alpha;
                    if (pr.bias) v += pr.bias[n];
                    if (pr.act == 1) v = gelu_exact(v);
                    C[(long long)m * pr.ldc + n] = v;
                }
            }
}

// ---------------- softmax over L for pScore (in place) ----------------
__global__ __launch_bounds__(256) void softmaxL_kernel(float* pS, const int* maskPAD) {
    const int b = blockIdx.x / NP, p = blockIdx.x % NP;
    const int tid = threadIdx.x;
    __shared__ float red[256];
    float v[4];
    #pragma unroll
    for (int i = 0; i < 4; ++i) {
        int l = tid + 256 * i;
        int pad = (maskPAD[(long long)b * NL * NL + l] == 0);
        float s = pS[((long long)b * NL + l) * NP + p];
        v[i] = pad ? -32768.0f : s;
    }
    float mx = fmaxf(fmaxf(v[0], v[1]), fmaxf(v[2], v[3]));
    red[tid] = mx; __syncthreads();
    for (int s = 128; s > 0; s >>= 1) {
        if (tid < s) red[tid] = fmaxf(red[tid], red[tid + s]);
        __syncthreads();
    }
    mx = red[0]; __syncthreads();
    float lsum = 0.0f;
    #pragma unroll
    for (int i = 0; i < 4; ++i) { v[i] = __expf(v[i] - mx); lsum += v[i]; }
    red[tid] = lsum; __syncthreads();
    for (int s = 128; s > 0; s >>= 1) {
        if (tid < s) red[tid] += red[tid + s];
        __syncthreads();
    }
    float inv = 1.0f / red[0];
    #pragma unroll
    for (int i = 0; i < 4; ++i) {
        int l = tid + 256 * i;
        pS[((long long)b * NL + l) * NP + p] = v[i] * inv;
    }
}

// ---------------- exclusive prefix sum of pAlpha over L ----------------
__global__ void prefix_kernel(const float* pA, float* PS) {
    int idx = blockIdx.x * blockDim.x + threadIdx.x;
    if (idx >= NB * NP) return;
    int b = idx / NP, p = idx % NP;
    const float* src = pA + (long long)b * NL * NP + p;
    float* dst = PS + (long long)b * (NL + 1) * NP + p;
    float accv = 0.0f;
    dst[0] = 0.0f;
    for (int m = 0; m < NL; ++m) {
        accv += src[(long long)m * NP];
        dst[(long long)(m + 1) * NP] = accv;
    }
}

// ---------------- relative-position bias via bucketed prefix sums ----------------
// tab(l,m) buckets by distance d=|l-m|: 0..6 singleton, 7:[7,8], 8:[9,10],
// 9:[11,14], 10:[15,22], 11:[23,38], 12:[39,70], 13:[71,134], 14:[135,1023]
__global__ __launch_bounds__(64) void relbias_kernel(const float* PS,
                                                     const float* embK, const float* embB,
                                                     float* Kb, float* bb) {
    const int dlo[15] = {0,1,2,3,4,5,6,7,9,11,15,23,39,71,135};
    const int dhi[15] = {0,1,2,3,4,5,6,8,10,14,22,38,70,134,1023};
    const int bl = blockIdx.x;
    const int b = bl / NL, l = bl % NL;
    const int p = threadIdx.x;
    const float* ps = PS + (long long)b * (NL + 1) * NP + p;

    float S[15];
    S[0] = ps[(long long)(l + 1) * NP] - ps[(long long)l * NP];
    #pragma unroll
    for (int d = 1; d < 15; ++d) {
        float s = 0.0f;
        int m2 = l - dlo[d];
        if (m2 >= 0) {
            int m1 = l - dhi[d]; if (m1 < 0) m1 = 0;
            s += ps[(long long)(m2 + 1) * NP] - ps[(long long)m1 * NP];
        }
        int m1r = l + dlo[d];
        if (m1r <= NL - 1) {
            int m2r = l + dhi[d]; if (m2r > NL - 1) m2r = NL - 1;
            s += ps[(long long)(m2r + 1) * NP] - ps[(long long)m1r * NP];
        }
        S[d] = s;
    }
    #pragma unroll
    for (int h = 0; h < NH; ++h) {
        float kv = 0.0f, bv = 0.0f;
        #pragma unroll
        for (int d = 0; d < 15; ++d) {
            kv += S[d] * embK[d * NH + h];
            bv += S[d] * embB[d * NH + h];
        }
        long long o = (((long long)b * NH + h) * NL + l) * NP + p;
        Kb[o] = kv;
        bb[o] = bv;
    }
}

// ---------------- attention softmax over P (one wave per row) ----------------
__global__ __launch_bounds__(256) void attn_softmax_kernel(float* scores,
                                                           const float* Kb, const float* bb) {
    const int row  = blockIdx.x * 8 + (threadIdx.x >> 5);
    const int lane = threadIdx.x & 31;
    const long long base = (long long)row * NP;
    float s0 = scores[base + lane]      * Kb[base + lane]      + bb[base + lane];
    float s1 = scores[base + lane + 32] * Kb[base + lane + 32] + bb[base + lane + 32];
    float mx = fmaxf(s0, s1);
    #pragma unroll
    for (int m = 16; m > 0; m >>= 1) mx = fmaxf(mx, __shfl_xor(mx, m, 32));
    float e0 = __expf(s0 - mx), e1 = __expf(s1 - mx);
    float sm = e0 + e1;
    #pragma unroll
    for (int m = 16; m > 0; m >>= 1) sm += __shfl_xor(sm, m, 32);
    float inv = 1.0f / sm;
    scores[base + lane]      = e0 * inv;
    scores[base + lane + 32] = e1 * inv;
}

// ---------------- residual add + LayerNorm over C=512 ----------------
__global__ __launch_bounds__(256) void ln_kernel(const float* xa, const float* xb,
                                                 const float* g, const float* bt, float* out) {
    const int row = blockIdx.x, tid = threadIdx.x;
    __shared__ float r1[256], r2[256];
    long long base = (long long)row * NC;
    float x0 = xa[base + tid]       + xb[base + tid];
    float x1 = xa[base + tid + 256] + xb[base + tid + 256];
    r1[tid] = x0 + x1;
    r2[tid] = x0 * x0 + x1 * x1;
    __syncthreads();
    for (int s = 128; s > 0; s >>= 1) {
        if (tid < s) { r1[tid] += r1[tid + s]; r2[tid] += r2[tid + s]; }
        __syncthreads();
    }
    float mu  = r1[0] * (1.0f / NC);
    float var = r2[0] * (1.0f / NC) - mu * mu;
    float rs  = rsqrtf(var + 1e-5f);
    out[base + tid]       = (x0 - mu) * rs * g[tid]       + bt[tid];
    out[base + tid + 256] = (x1 - mu) * rs * g[tid + 256] + bt[tid + 256];
}

// ---------------- host launch ----------------
extern "C" void kernel_launch(void* const* d_in, const int* in_sizes, int n_in,
                              void* d_out, int out_size, void* d_ws, size_t ws_size,
                              hipStream_t stream) {
    const float* qx  = (const float*)d_in[0];
    const float* kx  = (const float*)d_in[1];
    const float* vx  = (const float*)d_in[2];
    const int*   maskPAD = (const int*)d_in[3];
    const float* Wp  = (const float*)d_in[4];
    const float* bp  = (const float*)d_in[5];
    const float* WQ  = (const float*)d_in[6];
    const float* bQ  = (const float*)d_in[7];
    const float* WK  = (const float*)d_in[8];
    const float* bK  = (const float*)d_in[9];
    const float* WV  = (const float*)d_in[10];
    const float* bV  = (const float*)d_in[11];
    const float* WO  = (const float*)d_in[12];
    const float* bO  = (const float*)d_in[13];
    const float* embK = (const float*)d_in[14];
    const float* embB = (const float*)d_in[15];
    const float* ln1_g = (const float*)d_in[16];
    const float* ln1_b = (const float*)d_in[17];
    const float* ln2_g = (const float*)d_in[18];
    const float* ln2_b = (const float*)d_in[19];
    const float* W1  = (const float*)d_in[20];
    const float* b1  = (const float*)d_in[21];
    const float* W2  = (const float*)d_in[22];
    const float* b2  = (const float*)d_in[23];

    float* ws = (float*)d_ws;
    size_t o = 0;
    float* pAlpha = ws + o; o += (size_t)NB * NL * NP;
    float* PS     = ws + o; o += (size_t)NB * (NL + 1) * NP;
    float* kxp    = ws + o; o += (size_t)NB * NP * NC;
    float* vxp    = ws + o; o += (size_t)NB * NP * NC;
    float* kproj  = ws + o; o += (size_t)NB * NP * HDK;
    float* vproj  = ws + o; o += (size_t)NB * NP * HDK;
    float* qbuf   = ws + o; o += (size_t)NB * NL * HDK;       // q -> z -> ffnOut
    float* scores = ws + o; o += (size_t)NB * NH * NL * NP;   // scores -> ffn hidden
    float* Kb     = ws + o; o += (size_t)NB * NH * NL * NP;   // Kb -> attnOut
    float* bb     = ws + o; o += (size_t)NB * NH * NL * NP;   // bb -> z1

    auto gemm = [&](const float* A, const float* Bm, const float* bias, float* Cp,
                    int M, int N, int K, int lda, int ldb, int ldc,
                    int tA, int tB,
                    long long sA0, long long sA1, long long sB0, long long sB1,
                    long long sC0, long long sC1,
                    int batch, int subBatch, float alpha, int act) {
        GemmP p;
        p.A = A; p.Bm = Bm; p.bias = bias; p.C = Cp;
        p.M = M; p.N = N; p.K = K;
        p.lda = lda; p.ldb = ldb; p.ldc = ldc;
        p.sA0 = sA0; p.sA1 = sA1; p.sB0 = sB0; p.sB1 = sB1; p.sC0 = sC0; p.sC1 = sC1;
        p.subBatch = subBatch; p.alpha = alpha; p.act = act;
        dim3 g((M + BMT - 1) / BMT, (N + BNT - 1) / BNT, batch);
        if (tA == 0 && tB == 0)      gemm_wmma_f16<0, 0><<<g, dim3(256), 0, stream>>>(p);
        else if (tA == 1)            gemm_wmma_f16<1, 0><<<g, dim3(256), 0, stream>>>(p);
        else                         gemm_wmma_f16<0, 1><<<g, dim3(256), 0, stream>>>(p);
    };

    const long long sLP = (long long)NL * NP;
    const long long sLC = (long long)NL * NC;
    const long long sPC = (long long)NP * NC;
    const long long sHL = (long long)NH * NL * NP;

    // 1) pScore = vx @ Wp + bp
    gemm(vx, Wp, bp, pAlpha, NB * NL, NP, NC, NC, NP, NP, 0, 0,
         0, 0, 0, 0, 0, 0, 1, 1, 1.0f, 0);
    // 2) softmax over L (mask-aware, in place)
    softmaxL_kernel<<<NB * NP, 256, 0, stream>>>(pAlpha, maskPAD);
    // 3) prefix sums over L
    prefix_kernel<<<4, 256, 0, stream>>>(pAlpha, PS);
    // 4,5) kxp/vxp = pAlpha^T @ kx/vx
    gemm(pAlpha, kx, nullptr, kxp, NP, NC, NL, NP, NC, NC, 1, 0,
         sLP, 0, sLC, 0, sPC, 0, NB, 1, 1.0f, 0);
    gemm(pAlpha, vx, nullptr, vxp, NP, NC, NL, NP, NC, NC, 1, 0,
         sLP, 0, sLC, 0, sPC, 0, NB, 1, 1.0f, 0);
    // 6) q = qx @ WQ + bQ
    gemm(qx, WQ, bQ, qbuf, NB * NL, HDK, NC, NC, HDK, HDK, 0, 0,
         0, 0, 0, 0, 0, 0, 1, 1, 1.0f, 0);
    // 7,8) pooled k/v projections
    gemm(kxp, WK, bK, kproj, NB * NP, HDK, NC, NC, HDK, HDK, 0, 0,
         0, 0, 0, 0, 0, 0, 1, 1, 1.0f, 0);
    gemm(vxp, WV, bV, vproj, NB * NP, HDK, NC, NC, HDK, HDK, 0, 0,
         0, 0, 0, 0, 0, 0, 1, 1, 1.0f, 0);
    // 9) rel-pos bias
    relbias_kernel<<<NB * NL, 64, 0, stream>>>(PS, embK, embB, Kb, bb);
    // 10) scores = 0.125 * q @ k^T per (b,h)
    gemm(qbuf, kproj, nullptr, scores, NL, NP, NDK, HDK, HDK, NP, 0, 1,
         (long long)NL * HDK, 64, (long long)NP * HDK, 64, sHL, sLP,
         NB * NH, NH, 0.125f, 0);
    // 11) softmax_p(scores * Kb + bb)
    attn_softmax_kernel<<<NB * NH * NL / 8, 256, 0, stream>>>(scores, Kb, bb);
    // 12) z = alpha @ v -> qbuf (head-interleaved)
    gemm(scores, vproj, nullptr, qbuf, NL, NDK, NP, NP, HDK, HDK, 0, 0,
         sHL, sLP, (long long)NP * HDK, 64, (long long)NL * HDK, 64,
         NB * NH, NH, 1.0f, 0);
    // 13) attnOut = z @ WO + bO -> Kb
    gemm(qbuf, WO, bO, Kb, NB * NL, NC, HDK, HDK, NC, NC, 0, 0,
         0, 0, 0, 0, 0, 0, 1, 1, 1.0f, 0);
    // 14) z1 = LN1(vx + attnOut) -> bb
    ln_kernel<<<NB * NL, 256, 0, stream>>>(vx, Kb, ln1_g, ln1_b, bb);
    // 15) FFN in 4 row-chunks (hidden reuses scores buffer)
    for (int c = 0; c < 4; ++c) {
        const float* z1c = bb + (size_t)c * 4096 * NC;
        float* outc = qbuf + (size_t)c * 4096 * NC;
        gemm(z1c, W1, b1, scores, 4096, FF, NC, NC, FF, FF, 0, 0,
             0, 0, 0, 0, 0, 0, 1, 1, 1.0f, 1);
        gemm(scores, W2, b2, outc, 4096, NC, FF, FF, NC, NC, 0, 0,
             0, 0, 0, 0, 0, 0, 1, 1, 1.0f, 0);
    }
    // 16) out = LN2(z1 + ffnOut)
    ln_kernel<<<NB * NL, 256, 0, stream>>>(bb, qbuf, ln2_g, ln2_b, (float*)d_out);
}
